// TemporalMoEViT_27633819582948
// MI455X (gfx1250) — matrix operand order repaired and checked by reading, not compiled, not run on hardware
//
#include <hip/hip_runtime.h>
#include <hip/hip_bf16.h>

// ---------------- CDNA5 WMMA types ----------------
typedef __attribute__((ext_vector_type(16))) _Float16 v16h;
typedef __attribute__((ext_vector_type(8)))  _Float16 v8h;
typedef __attribute__((ext_vector_type(4)))  _Float16 v4h;
typedef __attribute__((ext_vector_type(8)))  float    v8f;

// ---------------- Model constants ----------------
#define Dm      768
#define HEADS   12
#define DH      64
#define NLAYERS 4
#define NEXP    8
#define FD      64
#define DDm     32
#define Bb      16
#define QLEN    32
#define IMGs    224
#define NPATCH  196
#define Ss      229           // 1 + 32 + 196
#define NT      3664          // Bb * Ss
#define H4      3072
#define NC      1000
#define MPATCH  3136          // Bb * NPATCH
#define KPATCH  6144          // 24*16*16

// GEMM tiling
#define BM 128
#define BN 64
#define BK 32
#define LSTR 40               // f16 LDS row stride (32 + 8 pad)

__device__ __forceinline__ float gelu_tanh(float x) {
    float x3 = x * x * x;
    return 0.5f * x * (1.f + tanhf(0.7978845608028654f * (x + 0.044715f * x3)));
}

// ---- WMMA fragment loads from LDS (layouts per CDNA5 ISA 7.12.2) ----
__device__ __forceinline__ v16h load_frag_A(const _Float16* sA, int row, int lane) {
    int kc = (lane >> 4) << 3;                 // 0 or 8
    v8h lo = *(const v8h*)(sA + row * LSTR + kc);        // K = kc..kc+7
    v8h hi = *(const v8h*)(sA + row * LSTR + kc + 16);   // K = kc+16..kc+23
    return __builtin_shufflevector(lo, hi, 0,1,2,3,4,5,6,7,8,9,10,11,12,13,14,15);
}
__device__ __forceinline__ v16h load_frag_B(const _Float16* sB, int col, int lane) {
    int kc = (lane >> 4) << 3;                 // 0 or 8
    v8h lo = *(const v8h*)(sB + col * LSTR + kc);
    v8h hi = *(const v8h*)(sB + col * LSTR + kc + 16);
    return __builtin_shufflevector(lo, hi, 0,1,2,3,4,5,6,7,8,9,10,11,12,13,14,15);
}
__device__ __forceinline__ void wmma_step(const _Float16* sA, const _Float16* sB,
                                          int lane, int wave, v8f acc[4]) {
    v16h af = load_frag_A(sA, wave * 16 + (lane & 15), lane);
#pragma unroll
    for (int cn = 0; cn < 4; ++cn) {
        v16h bf = load_frag_B(sB, cn * 16 + (lane & 15), lane);
        acc[cn] = __builtin_amdgcn_wmma_f32_16x16x32_f16(
            false, af, false, bf, (short)0, acc[cn], false, false);
    }
}

// ---------------- Generic WMMA GEMM:  C = epi(A[MxK] @ B[KxN] + bias) ----------------
// All call sites have Kd % 32 == 0 and 16B-aligned rows (lda/ldb multiples of 4).
// EPI 0: C = acc+bias      1: C = gelu(acc+bias)
// EPI 2: C += scale[r*sstride]*(acc+bias)      3: C = resid + acc + bias
template <int EPI>
__global__ __launch_bounds__(256)
void gemm_f16_wmma(const float* __restrict__ A, const float* __restrict__ Bm,
                   const float* __restrict__ bias, float* __restrict__ C,
                   const float* __restrict__ resid, const float* __restrict__ scale, int sstride,
                   int M, int N, int Kd, int lda, int ldb, int ldc) {
    __shared__ __align__(16) _Float16 sA[BM * LSTR];
    __shared__ __align__(16) _Float16 sB[BN * LSTR];
    const int tid = threadIdx.x, lane = tid & 31, wave = tid >> 5;
    const int rb0 = blockIdx.y * BM, nb0 = blockIdx.x * BN;

    v8f acc[4] = {};
    float4 ra[4], rbv[2];

    // Register-stage one K-tile: branchless clamped b128 loads.
    auto loadT = [&](int kb) {
#pragma unroll
        for (int i = 0; i < 4; ++i) {                  // A tile: 128x32 f32, 1024 float4s
            int vi = i * 256 + tid;
            int r = vi >> 3, k = (vi & 7) << 2;
            int gr = rb0 + r;
            int cr = gr < M ? gr : M - 1;              // clamp, no branch
            float4 t = *(const float4*)(A + (long)cr * lda + kb + k);
            if (gr >= M) t = float4{0.f, 0.f, 0.f, 0.f};
            ra[i] = t;
        }
#pragma unroll
        for (int i = 0; i < 2; ++i) {                  // B tile: 32x64 f32, 512 float4s
            int vi = i * 256 + tid;
            int kr = vi >> 4, nc = (vi & 15) << 2;
            int gk = kb + kr, gn = nb0 + nc;
            float4 t;
            if (gn + 3 < N) {
                t = *(const float4*)(Bm + (long)gk * ldb + gn);
            } else {                                   // N-edge (e.g. N=1000 head)
                const float* row = Bm + (long)gk * ldb;
                t.x = (gn + 0 < N) ? row[gn + 0] : 0.f;
                t.y = (gn + 1 < N) ? row[gn + 1] : 0.f;
                t.z = (gn + 2 < N) ? row[gn + 2] : 0.f;
                t.w = (gn + 3 < N) ? row[gn + 3] : 0.f;
            }
            rbv[i] = t;
        }
    };
    auto storeT = [&]() {
#pragma unroll
        for (int i = 0; i < 4; ++i) {
            int vi = i * 256 + tid;
            int r = vi >> 3, k = (vi & 7) << 2;
            v4h h = {(_Float16)ra[i].x, (_Float16)ra[i].y, (_Float16)ra[i].z, (_Float16)ra[i].w};
            *(v4h*)(sA + r * LSTR + k) = h;            // b64 store
        }
#pragma unroll
        for (int i = 0; i < 2; ++i) {                  // transpose to n-major
            int vi = i * 256 + tid;
            int kr = vi >> 4, nc = (vi & 15) << 2;
            sB[(nc + 0) * LSTR + kr] = (_Float16)rbv[i].x;
            sB[(nc + 1) * LSTR + kr] = (_Float16)rbv[i].y;
            sB[(nc + 2) * LSTR + kr] = (_Float16)rbv[i].z;
            sB[(nc + 3) * LSTR + kr] = (_Float16)rbv[i].w;
        }
    };

    loadT(0);
    for (int kb = 0; kb < Kd; kb += BK) {
        storeT();
        __syncthreads();
        if (kb + BK < Kd) loadT(kb + BK);              // overlap next-tile loads with WMMA
        if (kb + 2 * BK < Kd) {                        // gfx1250 global_prefetch_b8
            int pr = rb0 + (tid & 127); if (pr > M - 1) pr = M - 1;
            __builtin_prefetch(&A[(long)pr * lda + kb + 2 * BK], 0, 1);
        }
        wmma_step(sA, sB, lane, wave, acc);
        __syncthreads();
    }
#pragma unroll
    for (int cn = 0; cn < 4; ++cn) {
        int c = nb0 + cn * 16 + (lane & 15);
        if (c >= N) continue;
        float bv = bias ? bias[c] : 0.f;
#pragma unroll
        for (int v = 0; v < 8; ++v) {
            int r = rb0 + wave * 16 + ((lane >> 4) << 3) + v;
            if (r >= M) continue;
            float val = acc[cn][v] + bv;
            long o = (long)r * ldc + c;
            if constexpr (EPI == 1) val = gelu_tanh(val);
            if constexpr (EPI == 2)      C[o] += scale[(long)r * sstride] * val;
            else if constexpr (EPI == 3) C[o] = resid[o] + val;
            else                         C[o] = val;
        }
    }
}

// ---------------- Patch-embed GEMM (im2col gather, K=6144) ----------------
// A 4-wide chunk at k%4==0 stays inside one 16-wide patch row -> float4 gather is legal.
__global__ __launch_bounds__(256)
void patch_embed_wmma(const float* __restrict__ video, const float* __restrict__ pw,
                      const float* __restrict__ pb, float* __restrict__ x) {
    __shared__ __align__(16) _Float16 sA[BM * LSTR];
    __shared__ __align__(16) _Float16 sB[BN * LSTR];
    const int tid = threadIdx.x, lane = tid & 31, wave = tid >> 5;
    const int rb0 = blockIdx.y * BM, nb0 = blockIdx.x * BN;
    v8f acc[4] = {};
    float4 ra[4], rbv[2];

    auto loadT = [&](int kb) {
#pragma unroll
        for (int i = 0; i < 4; ++i) {
            int vi = i * 256 + tid;
            int r = vi >> 3, k = (vi & 7) << 2;
            int gr = rb0 + r;
            int cr = gr < MPATCH ? gr : MPATCH - 1;
            int gk = kb + k;
            int b = cr / NPATCH, p = cr % NPATCH;
            int py = p / 14, px = p % 14;
            int c = gk >> 8, ph = (gk >> 4) & 15, pwp = gk & 15;
            float4 t = *(const float4*)(video +
                (((long)(b * 24 + c) * IMGs) + py * 16 + ph) * IMGs + px * 16 + pwp);
            if (gr >= MPATCH) t = float4{0.f, 0.f, 0.f, 0.f};
            ra[i] = t;
        }
#pragma unroll
        for (int i = 0; i < 2; ++i) {                  // patch_w is [768][6144] (N-major, k contig)
            int vi = i * 256 + tid;
            int nc = vi >> 3, k = (vi & 7) << 2;
            rbv[i] = *(const float4*)(pw + (long)(nb0 + nc) * KPATCH + kb + k);
        }
    };
    auto storeT = [&]() {
#pragma unroll
        for (int i = 0; i < 4; ++i) {
            int vi = i * 256 + tid;
            int r = vi >> 3, k = (vi & 7) << 2;
            v4h h = {(_Float16)ra[i].x, (_Float16)ra[i].y, (_Float16)ra[i].z, (_Float16)ra[i].w};
            *(v4h*)(sA + r * LSTR + k) = h;
        }
#pragma unroll
        for (int i = 0; i < 2; ++i) {
            int vi = i * 256 + tid;
            int nc = vi >> 3, k = (vi & 7) << 2;
            v4h h = {(_Float16)rbv[i].x, (_Float16)rbv[i].y, (_Float16)rbv[i].z, (_Float16)rbv[i].w};
            *(v4h*)(sB + nc * LSTR + k) = h;
        }
    };

    loadT(0);
    for (int kb = 0; kb < KPATCH; kb += BK) {
        storeT();
        __syncthreads();
        if (kb + BK < KPATCH) loadT(kb + BK);
        wmma_step(sA, sB, lane, wave, acc);
        __syncthreads();
    }
#pragma unroll
    for (int cn = 0; cn < 4; ++cn) {
        int c = nb0 + cn * 16 + (lane & 15);
        if (c >= Dm) continue;
        float bv = pb[c];
#pragma unroll
        for (int v = 0; v < 8; ++v) {
            int r = rb0 + wave * 16 + ((lane >> 4) << 3) + v;
            if (r >= MPATCH) continue;
            int b = r / NPATCH, p = r % NPATCH;
            long tok = (long)b * Ss + 1 + QLEN + p;
            x[tok * Dm + c] = acc[cn][v] + bv;
        }
    }
}

// ---------------- Attention: wave per (b, h, q-row), softmax over register-held scores ----------------
__global__ __launch_bounds__(256)
void attn_kernel(const float* __restrict__ Q, const float* __restrict__ Kt,
                 const float* __restrict__ V, float* __restrict__ O) {
    __shared__ float qs[8][64];
    __shared__ float ps[8][232];
    const int lane = threadIdx.x & 31, wave = threadIdx.x >> 5;
    const int b = blockIdx.z, h = blockIdx.y;
    const int s0 = blockIdx.x * 8 + wave;
    if (s0 >= Ss) return;                      // wave-uniform exit; no block barriers used
    const long rowQ = ((long)b * Ss + s0) * Dm + h * DH;
    qs[wave][lane]      = Q[rowQ + lane] * 0.125f;       // dh^-0.5 folded into q
    qs[wave][lane + 32] = Q[rowQ + lane + 32] * 0.125f;

    float sc[8], m = -3.0e38f;
#pragma unroll
    for (int jj = 0; jj < 8; ++jj) {
        int j = jj * 32 + lane;
        float s = -3.0e38f;
        if (j < Ss) {
            const float* kp = Kt + ((long)b * Ss + j) * Dm + h * DH;
            float a = 0.f;
#pragma unroll
            for (int d = 0; d < DH; ++d) a += qs[wave][d] * kp[d];
            s = a;
        }
        sc[jj] = s;
        m = fmaxf(m, s);
    }
#pragma unroll
    for (int o = 16; o > 0; o >>= 1) m = fmaxf(m, __shfl_xor(m, o, 32));
    float den = 0.f;
#pragma unroll
    for (int jj = 0; jj < 8; ++jj) {
        int j = jj * 32 + lane;
        float e = (j < Ss) ? __expf(sc[jj] - m) : 0.f;
        sc[jj] = e; den += e;
    }
#pragma unroll
    for (int o = 16; o > 0; o >>= 1) den += __shfl_xor(den, o, 32);
    float inv = 1.f / den;
#pragma unroll
    for (int jj = 0; jj < 8; ++jj) {
        int j = jj * 32 + lane;
        if (j < 232) ps[wave][j] = sc[jj] * inv;
    }
    float a0 = 0.f, a1 = 0.f;
    const float* vb = V + (long)b * Ss * Dm + h * DH;
    for (int j = 0; j < Ss; ++j) {
        float p = ps[wave][j];
        a0 += p * vb[(long)j * Dm + lane];
        a1 += p * vb[(long)j * Dm + lane + 32];
    }
    O[rowQ + lane] = a0;
    O[rowQ + lane + 32] = a1;
}

// ---------------- LayerNorm: wave per row ----------------
__global__ __launch_bounds__(256)
void layernorm_kernel(const float* __restrict__ X, const float* __restrict__ g,
                      const float* __restrict__ bta, float* __restrict__ Y, int rows) {
    const int lane = threadIdx.x & 31, wave = threadIdx.x >> 5;
    const int r = blockIdx.x * 8 + wave;
    if (r >= rows) return;
    const float* xr = X + (long)r * Dm;
    float vbuf[24], s = 0.f;
#pragma unroll
    for (int i = 0; i < 24; ++i) { vbuf[i] = xr[lane + i * 32]; s += vbuf[i]; }
#pragma unroll
    for (int o = 16; o > 0; o >>= 1) s += __shfl_xor(s, o, 32);
    float mean = s * (1.f / Dm);
    float vs = 0.f;
#pragma unroll
    for (int i = 0; i < 24; ++i) { float d = vbuf[i] - mean; vs += d * d; }
#pragma unroll
    for (int o = 16; o > 0; o >>= 1) vs += __shfl_xor(vs, o, 32);
    float rinv = rsqrtf(vs * (1.f / Dm) + 1e-5f);
    float* yr = Y + (long)r * Dm;
#pragma unroll
    for (int i = 0; i < 24; ++i) {
        int c = lane + i * 32;
        yr[c] = (vbuf[i] - mean) * rinv * g[c] + bta[c];
    }
}

// ---------------- Router: thread per row (softmax / top-2 / renorm) ----------------
__global__ void router_kernel(const float* __restrict__ Xin, const float* __restrict__ rw,
                              const float* __restrict__ rb, float* __restrict__ combine,
                              float* __restrict__ probsOut, float* __restrict__ maskOut, int rows) {
    int r = blockIdx.x * 256 + threadIdx.x;
    if (r >= rows) return;
    const float* xr = Xin + (long)r * Dm;
    float lg[NEXP];
#pragma unroll
    for (int e = 0; e < NEXP; ++e) lg[e] = rb[e];
    for (int k = 0; k < Dm; ++k) {
        float xv = xr[k];
        const float* w = rw + (long)k * NEXP;
#pragma unroll
        for (int e = 0; e < NEXP; ++e) lg[e] += xv * w[e];
    }
    float m = lg[0];
#pragma unroll
    for (int e = 1; e < NEXP; ++e) m = fmaxf(m, lg[e]);
    float den = 0.f, p[NEXP];
#pragma unroll
    for (int e = 0; e < NEXP; ++e) { p[e] = __expf(lg[e] - m); den += p[e]; }
    float inv = 1.f / den;
#pragma unroll
    for (int e = 0; e < NEXP; ++e) p[e] *= inv;
    int i0 = 0;
#pragma unroll
    for (int e = 1; e < NEXP; ++e) if (p[e] > p[i0]) i0 = e;     // ties -> lower index
    int i1 = (i0 == 0) ? 1 : 0;
#pragma unroll
    for (int e = 0; e < NEXP; ++e) if (e != i0 && p[e] > p[i1]) i1 = e;
    float rsum = 1.f / (p[i0] + p[i1]);
#pragma unroll
    for (int e = 0; e < NEXP; ++e) {
        combine[(long)r * NEXP + e] = 0.f;
        maskOut[(long)r * NEXP + e] = 0.f;
        probsOut[(long)r * NEXP + e] = p[e];
    }
    combine[(long)r * NEXP + i0] = p[i0] * rsum;
    combine[(long)r * NEXP + i1] = p[i1] * rsum;
    maskOut[(long)r * NEXP + i0] = 1.f;
    maskOut[(long)r * NEXP + i1] = 1.f;
}

// ---------------- Deterministic aux-loss reduction (single block) ----------------
__global__ __launch_bounds__(256)
void aux_reduce_kernel(const float* __restrict__ probs, const float* __restrict__ mask,
                       float* __restrict__ auxAccum, int rows) {
    __shared__ float sp[256 * NEXP];
    __shared__ float sm[256 * NEXP];
    int tid = threadIdx.x;
    float ap[NEXP] = {}, am[NEXP] = {};
    for (int r = tid; r < rows; r += 256)
#pragma unroll
        for (int e = 0; e < NEXP; ++e) {
            ap[e] += probs[(long)r * NEXP + e];
            am[e] += mask[(long)r * NEXP + e];
        }
#pragma unroll
    for (int e = 0; e < NEXP; ++e) { sp[tid * NEXP + e] = ap[e]; sm[tid * NEXP + e] = am[e]; }
    __syncthreads();
    for (int stride = 128; stride > 0; stride >>= 1) {
        if (tid < stride)
#pragma unroll
            for (int e = 0; e < NEXP; ++e) {
                sp[tid * NEXP + e] += sp[(tid + stride) * NEXP + e];
                sm[tid * NEXP + e] += sm[(tid + stride) * NEXP + e];
            }
        __syncthreads();
    }
    if (tid == 0) {
        float invN = 1.f / rows, a = 0.f;
        for (int e = 0; e < NEXP; ++e) a += (sm[e] * invN) * (sp[e] * invN);
        auxAccum[0] += (float)NEXP * a;
    }
}

// ---------------- Small elementwise / gather kernels ----------------
__global__ void embed_kernel(const float* __restrict__ cls, const float* __restrict__ temb,
                             const int* __restrict__ qids, float* __restrict__ x) {
    int idx = blockIdx.x * 256 + threadIdx.x;
    if (idx >= Bb * Dm) return;
    int b = idx / Dm, d = idx % Dm;
    x[((long)b * Ss) * Dm + d] = cls[d];
    for (int i = 0; i < QLEN; ++i) {
        int id = qids[b * QLEN + i];
        x[((long)b * Ss + 1 + i) * Dm + d] = temb[(long)id * Dm + d];
    }
}
__global__ void avgq_kernel(const float* __restrict__ temb, const int* __restrict__ qids,
                            float* __restrict__ avgq) {
    int idx = blockIdx.x * 256 + threadIdx.x;
    if (idx >= Bb * Dm) return;
    int b = idx / Dm, d = idx % Dm;
    float s = 0.f;
    for (int i = 0; i < QLEN; ++i) s += temb[(long)qids[b * QLEN + i] * Dm + d];
    avgq[idx] = s * (1.f / QLEN);
}
__global__ void pos_add_kernel(const float* __restrict__ pos, float* __restrict__ x) {
    long idx = (long)blockIdx.x * 256 + threadIdx.x;
    if (idx >= (long)NT * Dm) return;
    int d = (int)(idx % Dm);
    int s = (int)((idx / Dm) % Ss);
    x[idx] += pos[(long)s * Dm + d];
}
__global__ void concat_kernel(const float* __restrict__ base, const float* __restrict__ extra,
                              float* __restrict__ out, int rows, int extraW, int bcast) {
    int din = Dm + extraW;
    long total = (long)rows * din;
    long idx = (long)blockIdx.x * 256 + threadIdx.x;
    if (idx >= total) return;
    long r = idx / din;
    int j = (int)(idx % din);
    float v;
    if (j < Dm) v = base[r * Dm + j];
    else {
        long er = bcast ? (r / Ss) : r;
        v = extra[er * extraW + (j - Dm)];
    }
    out[idx] = v;
}
__global__ void zero_kernel(float* p, long n) {
    long i = (long)blockIdx.x * 256 + threadIdx.x;
    if (i < n) p[i] = 0.f;
}
__global__ void add_kernel(float* __restrict__ x, const float* __restrict__ y, long n) {
    long i = (long)blockIdx.x * 256 + threadIdx.x;
    if (i < n) x[i] += y[i];
}
__global__ void extract_cls_kernel(const float* __restrict__ x, float* __restrict__ cb) {
    int idx = blockIdx.x * 256 + threadIdx.x;
    if (idx >= Bb * Dm) return;
    int b = idx / Dm, d = idx % Dm;
    cb[idx] = x[((long)b * Ss) * Dm + d];
}
__global__ void final_aux_kernel(const float* __restrict__ aux, float* __restrict__ out) {
    if (threadIdx.x == 0 && blockIdx.x == 0) out[Bb * NC] = aux[0] * (1.f / NLAYERS);
}

// ---------------- Param leaf indexing (JAX pytree: sorted dict keys) ----------------
// 0:cls 1:head_b 2:head_w 3..178:layers 179:patch_b 180:patch_w 181:pos 182:text_emb
// layer l base = 3 + l*44:
//   +0 bk +1 bo +2 bq +3 bv ; experts e: +4+e*4 {b1,b2,w1,w2}
//   +36 ln_b +37 ln_g +38 router_b +39 router_w +40 wk +41 wo +42 wq +43 wv
#define LF_CLS     0
#define LF_HEAD_B  1
#define LF_HEAD_W  2
#define LF_PATCH_B 179
#define LF_PATCH_W 180
#define LF_POS     181
#define LF_TEXT    182

extern "C" void kernel_launch(void* const* d_in, const int* in_sizes, int n_in,
                              void* d_out, int out_size, void* d_ws, size_t ws_size,
                              hipStream_t stream) {
    (void)in_sizes; (void)n_in; (void)out_size; (void)ws_size;
    const float* video  = (const float*)d_in[0];
    const int*   qids   = (const int*)d_in[1];
    const float* flow   = (const float*)d_in[2];
    const float* deltas = (const float*)d_in[3];
    const float* rawp   = (const float*)d_in[4];
    auto PRM = [&](int leaf) { return (const float*)d_in[5 + leaf]; };

    float* ws = (float*)d_ws;
    long off = 0;
    float* x    = ws + off; off += (long)NT * Dm;
    float* qb   = ws + off; off += (long)NT * Dm;
    float* kbf  = ws + off; off += (long)NT * Dm;
    float* vbf  = ws + off; off += (long)NT * Dm;
    float* ob   = ws + off; off += (long)NT * Dm;
    float* ffn  = ws + off; off += (long)NT * Dm;
    float* moe  = ws + off; off += (long)NT * Dm;
    float* hbuf = ws + off; off += (long)NT * H4;
    float* cbuf = ws + off; off += (long)NT * 1536;
    float* avgq = ws + off; off += (long)Bb * Dm;
    float* clsb = ws + off; off += (long)Bb * Dm;
    float* comb = ws + off; off += (long)NT * NEXP;
    float* prb  = ws + off; off += (long)NT * NEXP;
    float* msk  = ws + off; off += (long)NT * NEXP;
    float* auxw = ws + off; off += 8;

    dim3 blk(256);
    auto G1 = [](long n) { return dim3((unsigned)((n + 255) / 256)); };

    zero_kernel<<<G1(8), blk, 0, stream>>>(auxw, 8);
    embed_kernel<<<G1(Bb * Dm), blk, 0, stream>>>(PRM(LF_CLS), PRM(LF_TEXT), qids, x);
    avgq_kernel<<<G1(Bb * Dm), blk, 0, stream>>>(PRM(LF_TEXT), qids, avgq);
    {
        dim3 grid(Dm / BN, (MPATCH + BM - 1) / BM);
        patch_embed_wmma<<<grid, blk, 0, stream>>>(video, PRM(LF_PATCH_W), PRM(LF_PATCH_B), x);
    }
    pos_add_kernel<<<G1((long)NT * Dm), blk, 0, stream>>>(PRM(LF_POS), x);

    const int dinT[NEXP] = {Dm + FD, Dm + 768, Dm, 2 * Dm, Dm + DDm, Dm, Dm, Dm};
    const dim3 gD(Dm / BN, (NT + BM - 1) / BM);       // N=768 GEMMs over all tokens
    const dim3 gH(H4 / BN, (NT + BM - 1) / BM);       // N=3072 GEMMs

    for (int l = 0; l < NLAYERS; ++l) {
        int lb = 3 + l * 44;
        const float *wq = PRM(lb + 42), *bq = PRM(lb + 2);
        const float *wk = PRM(lb + 40), *bk = PRM(lb + 0);
        const float *wv = PRM(lb + 43), *bv = PRM(lb + 3);
        const float *wo = PRM(lb + 41), *bo = PRM(lb + 1);
        const float *lng = PRM(lb + 37), *lnb = PRM(lb + 36);
        const float *rwp = PRM(lb + 39), *rbp = PRM(lb + 38);

        gemm_f16_wmma<0><<<gD, blk, 0, stream>>>(x, wq, bq, qb,  nullptr, nullptr, 0, NT, Dm, Dm, Dm, Dm, Dm);
        gemm_f16_wmma<0><<<gD, blk, 0, stream>>>(x, wk, bk, kbf, nullptr, nullptr, 0, NT, Dm, Dm, Dm, Dm, Dm);
        gemm_f16_wmma<0><<<gD, blk, 0, stream>>>(x, wv, bv, vbf, nullptr, nullptr, 0, NT, Dm, Dm, Dm, Dm, Dm);
        attn_kernel<<<dim3((Ss + 7) / 8, HEADS, Bb), blk, 0, stream>>>(qb, kbf, vbf, ob);
        gemm_f16_wmma<3><<<gD, blk, 0, stream>>>(ob, wo, bo, x, x, nullptr, 0, NT, Dm, Dm, Dm, Dm, Dm);

        layernorm_kernel<<<dim3((NT + 7) / 8), blk, 0, stream>>>(x, lng, lnb, ffn, NT);
        router_kernel<<<G1(NT), blk, 0, stream>>>(ffn, rwp, rbp, comb, prb, msk, NT);
        aux_reduce_kernel<<<dim3(1), blk, 0, stream>>>(prb, msk, auxw, NT);
        zero_kernel<<<G1((long)NT * Dm), blk, 0, stream>>>(moe, (long)NT * Dm);

        for (int e = 0; e < NEXP; ++e) {
            int eb = lb + 4 + e * 4;
            const float *b1 = PRM(eb + 0), *b2 = PRM(eb + 1);
            const float *w1 = PRM(eb + 2), *w2 = PRM(eb + 3);
            int din = dinT[e];
            const float* Ae = ffn; int lda = Dm;
            if (e == 0) { concat_kernel<<<G1((long)NT * din), blk, 0, stream>>>(ffn, flow,   cbuf, NT, FD,  0); Ae = cbuf; lda = din; }
            else if (e == 1) { concat_kernel<<<G1((long)NT * din), blk, 0, stream>>>(ffn, rawp,   cbuf, NT, 768, 0); Ae = cbuf; lda = din; }
            else if (e == 3) { concat_kernel<<<G1((long)NT * din), blk, 0, stream>>>(ffn, avgq,   cbuf, NT, Dm,  1); Ae = cbuf; lda = din; }
            else if (e == 4) { concat_kernel<<<G1((long)NT * din), blk, 0, stream>>>(ffn, deltas, cbuf, NT, DDm, 0); Ae = cbuf; lda = din; }
            gemm_f16_wmma<1><<<gH, blk, 0, stream>>>(Ae, w1, b1, hbuf, nullptr, nullptr, 0, NT, H4, din, lda, H4, H4);
            gemm_f16_wmma<2><<<gD, blk, 0, stream>>>(hbuf, w2, b2, moe, nullptr, comb + e, NEXP, NT, Dm, H4, H4, Dm, Dm);
        }
        add_kernel<<<G1((long)NT * Dm), blk, 0, stream>>>(x, moe, (long)NT * Dm);
    }

    extract_cls_kernel<<<G1(Bb * Dm), blk, 0, stream>>>(x, clsb);
    {
        dim3 gh((NC + BN - 1) / BN, 1);
        gemm_f16_wmma<0><<<gh, blk, 0, stream>>>(clsb, PRM(LF_HEAD_W), PRM(LF_HEAD_B),
                                                 (float*)d_out, nullptr, nullptr, 0,
                                                 Bb, NC, Dm, Dm, NC, NC);
    }
    final_aux_kernel<<<dim3(1), dim3(1), 0, stream>>>(auxw, (float*)d_out);
}
